// GATNet2_6270652252168
// MI455X (gfx1250) — matrix operand order repaired and checked by reading, not compiled, not run on hardware
//
#include <hip/hip_runtime.h>
#include <hip/hip_bf16.h>

#define F_INC   256
#define HCC     256
#define HEADS   4
#define CHANS   64
#define NEG_SLOPE 0.2f
#define ENC_NEGINF 0x007FFFFFu

typedef __attribute__((ext_vector_type(16))) _Float16 v16h;
typedef __attribute__((ext_vector_type(8)))  float    v8f;

// ---- order-preserving float <-> u32 encoding for atomicMax-based segment max ----
__device__ __forceinline__ unsigned fenc(float f) {
    unsigned u = __float_as_uint(f);
    return (u & 0x80000000u) ? ~u : (u | 0x80000000u);
}
__device__ __forceinline__ float fdec(unsigned e) {
    unsigned b = (e & 0x80000000u) ? (e & 0x7fffffffu) : ~e;
    return __uint_as_float(b);
}

// ---- pack f32 weights B[K,Nc] -> f16 Bt[Nc,K] (K contiguous per column) ----
__global__ __launch_bounds__(256)
void pack_b_f16(const float* __restrict__ B, _Float16* __restrict__ Bt, int K, int Nc) {
    int idx = blockIdx.x * blockDim.x + threadIdx.x;
    if (idx >= K * Nc) return;
    int n = idx / K;
    int k = idx - n * K;
    Bt[idx] = (_Float16)B[(long)k * Nc + n];
}

// =====================================================================
// WMMA GEMM: D[M,Nc] = A[M,K] * B[K,Nc]  with B pre-packed f16 [Nc,K].
// One wave computes a 16x32 output strip (two 16x16 tiles), K-steps of 32.
// A layout (16x32 f16): lane L holds row M=L%16;
//   half i -> K = k0 + i + 16*(L/16)
//   (one contiguous 16-float chunk per lane -> four b128 loads)
// B layout (32x16 f16): lane L holds col N=L%16;
//   half i -> K = k0 + i + 16*(L/16)   (contiguous in Bt -> one 32B load)
// D layout (16x16 f32): lane L, vgpr r -> M = r + 8*(L/16), N = L%16
// Out-of-range rows are address-clamped; their results are never stored,
// so no predicated loads are needed (A row m only feeds D row m).
// =====================================================================
__global__ __launch_bounds__(256)
void gemm_wmma_f16(const float* __restrict__ A, const _Float16* __restrict__ Bt,
                   float* __restrict__ D, int M, int K, int Nc) {
    const int lane = threadIdx.x & 31;
    const int wave = threadIdx.x >> 5;
    const int wavesPerBlock = blockDim.x >> 5;
    const int tilesN2 = Nc >> 5;                  // pairs of 16-col tiles
    const int tilesM  = (M + 15) >> 4;
    const int tile = blockIdx.x * wavesPerBlock + wave;
    if (tile >= tilesM * tilesN2) return;         // uniform per wave
    const int tM = tile / tilesN2;
    const int tN = (tile - tM * tilesN2) << 1;    // first of two col tiles

    const int lid    = lane & 15;
    const int hiLane = lane >> 4;                 // 0 or 1
    int row = tM * 16 + lid;
    if (row >= M) row = M - 1;                    // clamp: garbage rows never stored
    const float4* __restrict__ arow4 = (const float4*)(A + (long)row * K);
    const int col0 = tN * 16 + lid;
    const int col1 = col0 + 16;
    const _Float16* __restrict__ bp0 = Bt + (long)col0 * K;
    const _Float16* __restrict__ bp1 = Bt + (long)col1 * K;

    v8f acc0 = {}, acc1 = {};
    for (int k0 = 0; k0 < K; k0 += 32) {
        // ---- A fragment: four b128 loads, convert to f16 ----
        const int base = (k0 >> 2) + (hiLane << 2);
        float4 c0 = arow4[base];
        float4 c1 = arow4[base + 1];
        float4 c2 = arow4[base + 2];
        float4 c3 = arow4[base + 3];
        v16h a;
        a[0]  = (_Float16)c0.x; a[1]  = (_Float16)c0.y;
        a[2]  = (_Float16)c0.z; a[3]  = (_Float16)c0.w;
        a[4]  = (_Float16)c1.x; a[5]  = (_Float16)c1.y;
        a[6]  = (_Float16)c1.z; a[7]  = (_Float16)c1.w;
        a[8]  = (_Float16)c2.x; a[9]  = (_Float16)c2.y;
        a[10] = (_Float16)c2.z; a[11] = (_Float16)c2.w;
        a[12] = (_Float16)c3.x; a[13] = (_Float16)c3.y;
        a[14] = (_Float16)c3.z; a[15] = (_Float16)c3.w;

        // ---- B fragments: 32B contiguous loads from packed layout ----
        const int kb = k0 + (hiLane << 4);
        v16h b0 = *(const v16h*)(bp0 + kb);
        v16h b1 = *(const v16h*)(bp1 + kb);

        acc0 = __builtin_amdgcn_wmma_f32_16x16x32_f16(
            false, a, false, b0, (short)0, acc0, false, false);
        acc1 = __builtin_amdgcn_wmma_f32_16x16x32_f16(
            false, a, false, b1, (short)0, acc1, false, false);
    }
#pragma unroll
    for (int r = 0; r < 8; ++r) {
        int mr = tM * 16 + r + (hiLane << 3);
        if (mr < M) {
            D[(long)mr * Nc + col0] = acc0[r];
            D[(long)mr * Nc + col1] = acc1[r];
        }
    }
}

// ---- a_src[n,h] = <h[n,h,:], att_src[h,:]>, same for a_dst ----
__global__ __launch_bounds__(256)
void attn_scores(const float* __restrict__ h, const float* __restrict__ attS,
                 const float* __restrict__ attD, float* __restrict__ aS,
                 float* __restrict__ aD, int N) {
    int idx = blockIdx.x * blockDim.x + threadIdx.x;
    if (idx >= N * HEADS) return;
    int hh = idx & (HEADS - 1);
    int n  = idx >> 2;
    const float* hp = h + (long)n * HCC + hh * CHANS;
    const float* sp = attS + hh * CHANS;
    const float* dp = attD + hh * CHANS;
    float s = 0.f, d = 0.f;
#pragma unroll 8
    for (int c = 0; c < CHANS; ++c) { float v = hp[c]; s += v * sp[c]; d += v * dp[c]; }
    aS[idx] = s; aD[idx] = d;
}

__global__ __launch_bounds__(256)
void fill_u32(unsigned* __restrict__ p, unsigned v, long n) {
    long i = (long)blockIdx.x * blockDim.x + threadIdx.x;
    if (i < n) p[i] = v;
}

__device__ __forceinline__ float edge_logit(const int* __restrict__ src,
                                            const int* __restrict__ dst,
                                            const float* __restrict__ aS,
                                            const float* __restrict__ aD,
                                            long e, int hh, int E, int& d) {
    int s;
    if (e < E) { s = src[e]; d = dst[e]; }
    else       { s = d = (int)(e - E); }            // self loop
    float l = aS[s * HEADS + hh] + aD[d * HEADS + hh];
    return l > 0.f ? l : NEG_SLOPE * l;
}

__global__ __launch_bounds__(256)
void edge_max(const int* __restrict__ src, const int* __restrict__ dst,
              const float* __restrict__ aS, const float* __restrict__ aD,
              unsigned* __restrict__ mEnc, int E, int N) {
    long idx = (long)blockIdx.x * blockDim.x + threadIdx.x;
    if (idx >= (long)(E + N) * HEADS) return;
    int hh = idx & (HEADS - 1); long e = idx >> 2; int d;
    float l = edge_logit(src, dst, aS, aD, e, hh, E, d);
    atomicMax(&mEnc[d * HEADS + hh], fenc(l));
}

__global__ __launch_bounds__(256)
void edge_denom(const int* __restrict__ src, const int* __restrict__ dst,
                const float* __restrict__ aS, const float* __restrict__ aD,
                const unsigned* __restrict__ mEnc, float* __restrict__ denom,
                int E, int N) {
    long idx = (long)blockIdx.x * blockDim.x + threadIdx.x;
    if (idx >= (long)(E + N) * HEADS) return;
    int hh = idx & (HEADS - 1); long e = idx >> 2; int d;
    float l = edge_logit(src, dst, aS, aD, e, hh, E, d);
    float m = fdec(mEnc[d * HEADS + hh]);
    atomicAdd(&denom[d * HEADS + hh], __expf(l - m));
}

__global__ __launch_bounds__(256)
void edge_alpha(const int* __restrict__ src, const int* __restrict__ dst,
                const float* __restrict__ aS, const float* __restrict__ aD,
                const unsigned* __restrict__ mEnc, const float* __restrict__ denom,
                float* __restrict__ alpha, int E, int N) {
    long idx = (long)blockIdx.x * blockDim.x + threadIdx.x;
    if (idx >= (long)(E + N) * HEADS) return;
    int hh = idx & (HEADS - 1); long e = idx >> 2; int d;
    float l = edge_logit(src, dst, aS, aD, e, hh, E, d);
    float m = fdec(mEnc[d * HEADS + hh]);
    alpha[idx] = __expf(l - m) / (denom[d * HEADS + hh] + 1e-16f);
}

// one thread per (edge, head, 4-channel chunk): gather float4, 4 atomic adds
__global__ __launch_bounds__(256)
void edge_scatter(const int* __restrict__ src, const int* __restrict__ dst,
                  const float* __restrict__ alpha, const float* __restrict__ h,
                  float* __restrict__ out, int E, int N) {
    long idx = (long)blockIdx.x * blockDim.x + threadIdx.x;
    if (idx >= (long)(E + N) * HEADS * (CHANS / 4)) return;
    int chunk = idx & 15;
    int hh    = (idx >> 4) & (HEADS - 1);
    long e    = idx >> 6;
    int s, d;
    if (e < E) { s = src[e]; d = dst[e]; } else { s = d = (int)(e - E); }
    float al = alpha[e * HEADS + hh];
    const float4 v = *(const float4*)(h + (long)s * HCC + hh * CHANS + chunk * 4);
    float* op = out + (long)d * HCC + hh * CHANS + chunk * 4;
    atomicAdd(op + 0, al * v.x);
    atomicAdd(op + 1, al * v.y);
    atomicAdd(op + 2, al * v.z);
    atomicAdd(op + 3, al * v.w);
}

__global__ __launch_bounds__(256)
void bias_elu(float* __restrict__ p, const float* __restrict__ b, long n) {
    long i = (long)blockIdx.x * blockDim.x + threadIdx.x;
    if (i >= n) return;
    float v = p[i] + b[i & (HCC - 1)];
    p[i] = v > 0.f ? v : (__expf(v) - 1.0f);
}

__global__ __launch_bounds__(256)
void pool_max(const float* __restrict__ h, const int* __restrict__ batch,
              unsigned* __restrict__ poolEnc, int N) {
    long idx = (long)blockIdx.x * blockDim.x + threadIdx.x;
    if (idx >= (long)N * HCC) return;
    int j = idx & (HCC - 1);
    int n = (int)(idx >> 8);
    atomicMax(&poolEnc[(long)batch[n] * HCC + j], fenc(h[idx]));
}

__global__ __launch_bounds__(256)
void pool_dec(const unsigned* __restrict__ poolEnc, float* __restrict__ pooled, int n) {
    int i = blockIdx.x * blockDim.x + threadIdx.x;
    if (i >= n) return;
    unsigned e = poolEnc[i];
    pooled[i] = (e == ENC_NEGINF) ? 0.0f : fdec(e);   // empty-graph guard
}

__global__ __launch_bounds__(256)
void add_bias(float* __restrict__ p, const float* __restrict__ b, int n, int ld) {
    int i = blockIdx.x * blockDim.x + threadIdx.x;
    if (i < n) p[i] += b[i % ld];
}

__global__ __launch_bounds__(256)
void lin2_kernel(const float* __restrict__ z1, const float* __restrict__ w,
                 const float* __restrict__ b, float* __restrict__ out, int G) {
    int i = blockIdx.x * blockDim.x + threadIdx.x;
    if (i >= G * 2) return;
    int g = i >> 1, c = i & 1;
    float s = b[c];
#pragma unroll 8
    for (int k = 0; k < 128; ++k) s += z1[g * 128 + k] * w[k * 2 + c];
    out[i] = s;
}

static inline int cdiv(long a, long b) { return (int)((a + b - 1) / b); }

extern "C" void kernel_launch(void* const* d_in, const int* in_sizes, int n_in,
                              void* d_out, int out_size, void* d_ws, size_t ws_size,
                              hipStream_t stream) {
    const float* x   = (const float*)d_in[0];
    const int*   ei  = (const int*)d_in[1];
    const int*   bat = (const int*)d_in[2];
    const float* W1  = (const float*)d_in[3];
    const float* as1 = (const float*)d_in[4];
    const float* ad1 = (const float*)d_in[5];
    const float* b1  = (const float*)d_in[6];
    const float* W2  = (const float*)d_in[7];
    const float* as2 = (const float*)d_in[8];
    const float* ad2 = (const float*)d_in[9];
    const float* b2  = (const float*)d_in[10];
    const float* l1w = (const float*)d_in[11];
    const float* l1b = (const float*)d_in[12];
    const float* l2w = (const float*)d_in[13];
    const float* l2b = (const float*)d_in[14];
    float* out = (float*)d_out;

    const int N = in_sizes[0] / F_INC;
    const int E = in_sizes[1] / 2;
    const int G = out_size / 2;
    const int* src = ei;
    const int* dst = ei + E;
    const long EN = (long)E + N;

    // ---- workspace carve-up ----
    char* wp = (char*)d_ws;
    auto carve = [&](size_t bytes) -> void* {
        void* p = (void*)wp; wp += (bytes + 255) & ~size_t(255); return p;
    };
    float*     buf0    = (float*)    carve((size_t)N * HCC * 4);
    float*     buf1    = (float*)    carve((size_t)N * HCC * 4);
    float*     aS      = (float*)    carve((size_t)N * HEADS * 4);
    float*     aD      = (float*)    carve((size_t)N * HEADS * 4);
    unsigned*  mEnc    = (unsigned*) carve((size_t)N * HEADS * 4);
    float*     denom   = (float*)    carve((size_t)N * HEADS * 4);
    float*     alpha   = (float*)    carve((size_t)EN * HEADS * 4);
    unsigned*  poolEnc = (unsigned*) carve((size_t)G * HCC * 4);
    float*     pooled  = (float*)    carve((size_t)G * HCC * 4);
    float*     z1      = (float*)    carve((size_t)G * (HCC / 2) * 4);
    _Float16*  Bt      = (_Float16*) carve((size_t)HCC * HCC * 2);   // packed weights

    const dim3 B(256);

    // ================= GAT layer (runs twice) =================
    for (int layer = 0; layer < 2; ++layer) {
        const float* hin  = (layer == 0) ? x   : buf1;
        const float* W    = (layer == 0) ? W1  : W2;
        const float* attS = (layer == 0) ? as1 : as2;
        const float* attD = (layer == 0) ? ad1 : ad2;
        const float* bias = (layer == 0) ? b1  : b2;

        // h = hin @ W  (WMMA f16, B pre-packed)
        pack_b_f16<<<cdiv(HCC * HCC, 256), B, 0, stream>>>(W, Bt, HCC, HCC);
        int tiles = cdiv(N, 16) * (HCC / 32);
        gemm_wmma_f16<<<cdiv(tiles, 8), B, 0, stream>>>(hin, Bt, buf0, N, HCC, HCC);

        attn_scores<<<cdiv((long)N * HEADS, 256), B, 0, stream>>>(buf0, attS, attD, aS, aD, N);

        fill_u32<<<cdiv((long)N * HEADS, 256), B, 0, stream>>>(mEnc, ENC_NEGINF, (long)N * HEADS);
        fill_u32<<<cdiv((long)N * HEADS, 256), B, 0, stream>>>((unsigned*)denom, 0u, (long)N * HEADS);
        fill_u32<<<cdiv((long)N * HCC, 256), B, 0, stream>>>((unsigned*)buf1, 0u, (long)N * HCC);

        edge_max   <<<cdiv(EN * HEADS, 256), B, 0, stream>>>(src, dst, aS, aD, mEnc, E, N);
        edge_denom <<<cdiv(EN * HEADS, 256), B, 0, stream>>>(src, dst, aS, aD, mEnc, denom, E, N);
        edge_alpha <<<cdiv(EN * HEADS, 256), B, 0, stream>>>(src, dst, aS, aD, mEnc, denom, alpha, E, N);
        edge_scatter<<<cdiv(EN * HEADS * (CHANS / 4), 256), B, 0, stream>>>(src, dst, alpha, buf0, buf1, E, N);

        bias_elu<<<cdiv((long)N * HCC, 256), B, 0, stream>>>(buf1, bias, (long)N * HCC);
    }

    // ================= global max pool =================
    fill_u32<<<cdiv((long)G * HCC, 256), B, 0, stream>>>(poolEnc, ENC_NEGINF, (long)G * HCC);
    pool_max<<<cdiv((long)N * HCC, 256), B, 0, stream>>>(buf1, bat, poolEnc, N);
    pool_dec<<<cdiv((long)G * HCC, 256), B, 0, stream>>>(poolEnc, pooled, G * HCC);

    // ================= MLP head =================
    pack_b_f16<<<cdiv(HCC * (HCC / 2), 256), B, 0, stream>>>(l1w, Bt, HCC, HCC / 2);
    {
        int tiles = cdiv(G, 16) * ((HCC / 2) / 32);
        gemm_wmma_f16<<<cdiv(tiles, 8), B, 0, stream>>>(pooled, Bt, z1, G, HCC, HCC / 2);
    }
    add_bias<<<cdiv((long)G * (HCC / 2), 256), B, 0, stream>>>(z1, l1b, G * (HCC / 2), HCC / 2);
    lin2_kernel<<<cdiv((long)G * 2, 256), B, 0, stream>>>(z1, l2w, l2b, out, G);
}
